// Attn_fusion_Block_36773509988780
// MI455X (gfx1250) — matrix operand order repaired and checked by reading, not compiled, not run on hardware
//
#include <hip/hip_runtime.h>

// ---------------------------------------------------------------------------
// Fused add + QKV proj + cross-head attention + output proj for MI455X
// (gfx1250, wave32, WMMA bf16 16x16x32 with f32 accumulate).
//
// Per workgroup: one tile of 64 tokens x 256 channels kept entirely in LDS.
//   Stage 1: x = xc + xt  -> bf16 LDS tile  Xs[t][c]        (coalesced HBM read)
//   Stage 2: Q,K,V = Xs @ W^T + b   (strip-mined WMMA: 1 B-frag -> 4 WMMAs)
//   Stage 3: per-token 8x8 cross-head attention (VALU + v_exp_f32 softmax)
//   Stage 4: Out^T[c][t] = O @ Wo^T + bo via WMMA, f32 in LDS
//   Stage 5: coalesced NCHW store
// ---------------------------------------------------------------------------

#define NT     256            // threads per block = 8 waves (wave32)
#define C_     256
#define NH     8
#define HD     32
#define HW     16384          // H*W
#define T_     64             // tokens per workgroup tile
#define PITCH  264            // bf16 row pitch (multiple of 8 -> 16B-aligned b128 loads)
#define SLOT   (T_ * PITCH * 2)        // 33792 bytes per LDS tile buffer
#define PO     65             // f32 pitch for OutS[c][t] (65 % 64 == 1 -> conflict-free)
#define LDS_BYTES (4 * SLOT)  // 135168 B  (well under 320 KB / WGP)

typedef __attribute__((ext_vector_type(16))) __bf16 bf16x16;
typedef __attribute__((ext_vector_type(8)))  float  f32x8;

union FragU { bf16x16 v; uint4 q[2]; };

__device__ __forceinline__ unsigned short f2bf(float f) {
  union { float f; unsigned u; } x; x.f = f;
  unsigned r = x.u + 0x7FFFu + ((x.u >> 16) & 1u);   // round-to-nearest-even
  return (unsigned short)(r >> 16);
}
__device__ __forceinline__ float bf2f(unsigned short h) {
  union { unsigned u; float f; } x; x.u = ((unsigned)h) << 16;
  return x.f;
}

// --------------------------- weight pre-pack (f32 -> bf16) ------------------
__global__ void cvt_weights(const float* __restrict__ wq, const float* __restrict__ wk,
                            const float* __restrict__ wv, const float* __restrict__ wo,
                            unsigned short* __restrict__ dst) {
  int idx = blockIdx.x * blockDim.x + threadIdx.x;     // 0 .. 4*65536-1
  if (idx >= 4 * 65536) return;
  int m = idx >> 16, i = idx & 65535;
  const float* s = (m == 0) ? wq : (m == 1) ? wk : (m == 2) ? wv : wo;
  dst[idx] = f2bf(s[i]);
}

// --------------------------- strip GEMM --------------------------------------
// One wave computes a full column strip: D[0..63 tokens][16 couts] =
// A(LDS token-major bf16, pitch PITCH) x W^T.  W row-major [cout][cin] bf16 is
// exactly the column-major B operand, so both fragment types are contiguous
// loads.  k-loop outer: each B fragment (2 global b128) feeds 4 WMMAs.
__device__ __forceinline__ void strip_gemm(const unsigned short* __restrict__ Abase,
                                           const unsigned short* __restrict__ Wbf,
                                           int tc, int lane, f32x8 acc[4]) {
  const int n  = lane & 15;       // A row (M=t) index == B column (N=cout) index
  const int hi = lane >> 4;       // lane-half selects contiguous K half 16*hi..16*hi+15
  const unsigned short* arow = Abase + n * PITCH;
  const unsigned short* brow = Wbf + (tc * 16 + n) * C_;
#pragma unroll
  for (int k = 0; k < 8; ++k) {           // K = 256 in 8 steps of 32
    const int kb = k * 32;
    FragU b;
    // B 32x16 bf16: lane<16 holds K 0..15 of its column, lane>=16 holds K 16..31
    b.q[0] = *(const uint4*)(brow + kb + hi * 16);
    b.q[1] = *(const uint4*)(brow + kb + hi * 16 + 8);
#pragma unroll
    for (int tr = 0; tr < 4; ++tr) {
      FragU a;
      // A 16x32 bf16: lane<16 holds K 0..15; lane>=16 holds K 16..31
      const unsigned short* ar = arow + tr * 16 * PITCH + kb;
      a.q[0] = *(const uint4*)(ar + hi * 16);
      a.q[1] = *(const uint4*)(ar + hi * 16 + 8);
      acc[tr] = __builtin_amdgcn_wmma_f32_16x16x32_bf16(
          /*neg_a=*/false, a.v, /*neg_b=*/false, b.v,
          /*c_mod=*/(short)0, acc[tr], /*reuse_a=*/false, /*reuse_b=*/false);
    }
  }
}

// --------------------------- fused attention block --------------------------
__global__ __launch_bounds__(NT, 1)
void attn_fused(const float* __restrict__ xc, const float* __restrict__ xt,
                const unsigned short* __restrict__ wbf,   // [Wq|Wk|Wv|Wo] bf16
                const float* __restrict__ bq, const float* __restrict__ bk,
                const float* __restrict__ bv, const float* __restrict__ bo,
                float* __restrict__ out) {
  extern __shared__ char smem[];
  unsigned short* bufA = (unsigned short*)smem;             // Xs, later Os
  unsigned short* Qs   = (unsigned short*)(smem + SLOT);
  unsigned short* Ks   = (unsigned short*)(smem + 2 * SLOT);
  unsigned short* Vs   = (unsigned short*)(smem + 3 * SLOT);
  float*          OutS = (float*)(smem + SLOT);             // overlays dead Qs+Ks

  const int tid  = threadIdx.x;
  const int lane = tid & 31;
  const int wave = tid >> 5;
  const int wg   = blockIdx.x;
  const int b    = wg >> 8;                 // 256 token tiles per batch image
  const int t0   = (wg & 255) * T_;
  const size_t gbase = (size_t)b * C_ * HW + t0;

  // ---- Stage 1: x = xc + xt -> bf16 LDS tile [t][c] (transposed on store)
  for (int i = tid; i < C_ * T_; i += NT) {
    int cin = i >> 6;                 // / T_
    int t   = i & (T_ - 1);
    size_t g = gbase + (size_t)cin * HW + t;
    bufA[t * PITCH + cin] = f2bf(xc[g] + xt[g]);
  }
  __syncthreads();

  // ---- Stage 2: Q,K,V (3 matrices x 16 cout strips = 48 strips over 8 waves)
  for (int strip = wave; strip < 48; strip += 8) {
    const int mat = strip >> 4;             // 0=Q 1=K 2=V
    const int tc  = strip & 15;
    f32x8 acc[4] = {{0.f,0.f,0.f,0.f,0.f,0.f,0.f,0.f},
                    {0.f,0.f,0.f,0.f,0.f,0.f,0.f,0.f},
                    {0.f,0.f,0.f,0.f,0.f,0.f,0.f,0.f},
                    {0.f,0.f,0.f,0.f,0.f,0.f,0.f,0.f}};
    strip_gemm(bufA, wbf + mat * 65536, tc, lane, acc);
    const float* bias   = (mat == 0) ? bq : (mat == 1) ? bk : bv;
    unsigned short* dst = (mat == 0) ? Qs : (mat == 1) ? Ks : Vs;
    const int n = lane & 15, hi = lane >> 4;
    const float bval = bias[tc * 16 + n];
#pragma unroll
    for (int tr = 0; tr < 4; ++tr)
#pragma unroll
      for (int r = 0; r < 8; ++r) {         // C/D layout: M = r + 8*hi, N = n
        const int t = tr * 16 + hi * 8 + r;
        dst[t * PITCH + tc * 16 + n] = f2bf(acc[tr][r] + bval);
      }
  }
  __syncthreads();

  // ---- Stage 3: per-token 8x8 cross-head attention (channel c = d*8 + h)
  if (tid < T_) {
    const int t = tid;
    const unsigned short* qrow = Qs + t * PITCH;
    const unsigned short* krow = Ks + t * PITCH;
    const unsigned short* vrow = Vs + t * PITCH;
    unsigned short* orow = bufA + t * PITCH;     // Os reuses Xs buffer

    float dot[NH][NH];
#pragma unroll
    for (int h = 0; h < NH; ++h)
#pragma unroll
      for (int g = 0; g < NH; ++g) dot[h][g] = 0.f;

    for (int d = 0; d < HD; ++d) {
      uint4 q4 = *(const uint4*)(qrow + d * 8);
      uint4 k4 = *(const uint4*)(krow + d * 8);
      const unsigned short* qh = (const unsigned short*)&q4;
      const unsigned short* kh = (const unsigned short*)&k4;
      float qf[8], kf[8];
#pragma unroll
      for (int h = 0; h < 8; ++h) { qf[h] = bf2f(qh[h]); kf[h] = bf2f(kh[h]); }
#pragma unroll
      for (int h = 0; h < 8; ++h)
#pragma unroll
        for (int g = 0; g < 8; ++g) dot[h][g] += qf[h] * kf[g];
    }
    const float scale = 0.17677669529663687f;    // 32^-0.5
#pragma unroll
    for (int h = 0; h < 8; ++h) {
      float m = -3.4e38f;
#pragma unroll
      for (int g = 0; g < 8; ++g) { dot[h][g] *= scale; m = fmaxf(m, dot[h][g]); }
      float s = 0.f;
#pragma unroll
      for (int g = 0; g < 8; ++g) { dot[h][g] = __expf(dot[h][g] - m); s += dot[h][g]; }
      const float inv = 1.f / s;
#pragma unroll
      for (int g = 0; g < 8; ++g) dot[h][g] *= inv;
    }
    for (int d = 0; d < HD; ++d) {
      uint4 v4 = *(const uint4*)(vrow + d * 8);
      const unsigned short* vh = (const unsigned short*)&v4;
      float vf[8];
#pragma unroll
      for (int g = 0; g < 8; ++g) vf[g] = bf2f(vh[g]);
#pragma unroll
      for (int h = 0; h < 8; ++h) {
        float o = 0.f;
#pragma unroll
        for (int g = 0; g < 8; ++g) o += dot[h][g] * vf[g];
        orow[h * HD + d] = f2bf(o);              // concat channel c' = h*32 + d
      }
    }
  }
  __syncthreads();

  // ---- Stage 4: Out^T[c][t] = O @ Wo^T + bo  (16 strips over 8 waves)
  for (int tc = wave; tc < 16; tc += 8) {
    f32x8 acc[4] = {{0.f,0.f,0.f,0.f,0.f,0.f,0.f,0.f},
                    {0.f,0.f,0.f,0.f,0.f,0.f,0.f,0.f},
                    {0.f,0.f,0.f,0.f,0.f,0.f,0.f,0.f},
                    {0.f,0.f,0.f,0.f,0.f,0.f,0.f,0.f}};
    strip_gemm(bufA, wbf + 3 * 65536, tc, lane, acc);
    const int n = lane & 15, hi = lane >> 4;
    const int c = tc * 16 + n;
    const float bval = bo[c];
#pragma unroll
    for (int tr = 0; tr < 4; ++tr)
#pragma unroll
      for (int r = 0; r < 8; ++r) {
        const int t = tr * 16 + hi * 8 + r;
        OutS[c * PO + t] = acc[tr][r] + bval;
      }
  }
  __syncthreads();

  // ---- Stage 5: coalesced store to NCHW output
  for (int i = tid; i < C_ * T_; i += NT) {
    int c = i >> 6, t = i & (T_ - 1);
    out[gbase + (size_t)c * HW + t] = OutS[c * PO + t];
  }
}

// ---------------------------------------------------------------------------
extern "C" void kernel_launch(void* const* d_in, const int* in_sizes, int n_in,
                              void* d_out, int out_size, void* d_ws, size_t ws_size,
                              hipStream_t stream) {
  const float* xc = (const float*)d_in[0];
  const float* xt = (const float*)d_in[1];
  const float* Wq = (const float*)d_in[2];
  const float* bq = (const float*)d_in[3];
  const float* Wk = (const float*)d_in[4];
  const float* bk = (const float*)d_in[5];
  const float* Wv = (const float*)d_in[6];
  const float* bv = (const float*)d_in[7];
  const float* Wo = (const float*)d_in[8];
  const float* bo = (const float*)d_in[9];
  float* out = (float*)d_out;
  unsigned short* wbf = (unsigned short*)d_ws;   // 512 KB of bf16 weights

  (void)hipFuncSetAttribute((const void*)attn_fused,
                            hipFuncAttributeMaxDynamicSharedMemorySize, LDS_BYTES);

  cvt_weights<<<(4 * 65536 + NT - 1) / NT, NT, 0, stream>>>(Wq, Wk, Wv, Wo, wbf);
  // 8 batches * 256 tiles of 64 tokens = 2048 workgroups
  attn_fused<<<2048, NT, LDS_BYTES, stream>>>(xc, xt, wbf, bq, bk, bv, bo, out);
}
